// AttentionHead_44375602102850
// MI455X (gfx1250) — matrix operand order repaired and checked by reading, not compiled, not run on hardware
//
#include <hip/hip_runtime.h>
#include <hip/hip_bf16.h>

// ---------------------------------------------------------------------------
// AttentionHead (no softmax) on MI455X / gfx1250, wave32 + WMMA.
//   output = (Q K^T * scale) V  ==  Q (K^T V) * scale      (pure linear chain)
// B=2, S=4096, D=1024, E=128.
// ---------------------------------------------------------------------------

#define B_   2
#define S_   4096
#define D_   1024
#define E_   128
#define SCALE_ 0.08838834764831845f   // 1/sqrt(128)

typedef _Float16 h8   __attribute__((ext_vector_type(8)));
typedef _Float16 v16h __attribute__((ext_vector_type(16)));
typedef float    f8   __attribute__((ext_vector_type(8)));
typedef float    v8f  __attribute__((ext_vector_type(8)));

// ---- WMMA fragment loaders (gfx1250 wave32 layouts, ISA §7.12.2) -----------
// A-matrix 16x32 f16: lanes 0-15 -> M=lane, K={kh..kh+7, 16+kh..16+kh+7},
// kh = (lane>>4)*8. Two contiguous 16B loads per lane.
__device__ __forceinline__ v16h load_a_f16(const _Float16* base, int stride, int k0) {
    int lane = threadIdx.x & 31;
    const _Float16* p = base + (size_t)(lane & 15) * stride + k0 + ((lane >> 4) * 8);
    h8 lo = *(const h8*)(p);
    h8 hi = *(const h8*)(p + 16);
    v16h out;
    #pragma unroll
    for (int i = 0; i < 8; ++i) { out[i] = lo[i]; out[i + 8] = hi[i]; }
    return out;
}

// Same fragment but source is f32 (x): load 2x 32B, convert to f16 in-register.
__device__ __forceinline__ v16h load_a_f32cvt(const float* base, int stride, int k0) {
    int lane = threadIdx.x & 31;
    const float* p = base + (size_t)(lane & 15) * stride + k0 + ((lane >> 4) * 8);
    f8 lo = *(const f8*)(p);
    f8 hi = *(const f8*)(p + 16);
    v16h out;
    #pragma unroll
    for (int i = 0; i < 8; ++i) { out[i] = (_Float16)lo[i]; out[i + 8] = (_Float16)hi[i]; }
    return out;
}

// B-matrix 32x16 f16: lane n=lane&15 holds column n, K={kh..kh+7, 16+kh..16+kh+7},
// kh = (lane>>4)*8 (same order as A). Source is row-major B^T (i.e. rows indexed by N),
// so one lane does two contiguous 16B loads.
__device__ __forceinline__ v16h load_b_f16(const _Float16* base, int stride, int k0) {
    int lane = threadIdx.x & 31;
    const _Float16* p = base + (size_t)(lane & 15) * stride + k0 + ((lane >> 4) * 8);
    h8 lo = *(const h8*)(p);
    h8 hi = *(const h8*)(p + 16);
    v16h out;
    #pragma unroll
    for (int i = 0; i < 8; ++i) { out[i] = lo[i]; out[i + 8] = hi[i]; }
    return out;
}

// ---- Phase 0: weights f32 -> f16 ------------------------------------------
__global__ __launch_bounds__(256) void conv_w(const float* __restrict__ qw,
                                              const float* __restrict__ kw,
                                              const float* __restrict__ vw,
                                              _Float16* __restrict__ Wh) {
    int i = blockIdx.x * 256 + threadIdx.x;            // 0 .. E_*D_-1
    Wh[i]                = (_Float16)qw[i];
    Wh[i +     E_ * D_]  = (_Float16)kw[i];
    Wh[i + 2 * E_ * D_]  = (_Float16)vw[i];
}

__global__ __launch_bounds__(256) void zero_m(float* __restrict__ M) {
    M[blockIdx.x * 256 + threadIdx.x] = 0.0f;          // B_*E_*E_ = 32768
}

// ---- Phase 1: fused QKV projection ----------------------------------------
// 512 blocks (= B*S/16), 8 waves/block; wave w owns E-cols [16w,16w+16) of a
// 16-seq-row tile. One A fragment (x, f32->f16) feeds 3 WMMAs per k-step.
// Q stored row-major f16 [B][S][E]; K,V stored transposed f16 [B][E][S]
// (packed b128 stores; makes phase-2/3 operand loads contiguous).
__global__ __launch_bounds__(256) void qkv_proj(const float* __restrict__ x,
                                                const _Float16* __restrict__ Wh,
                                                const float* __restrict__ qb,
                                                const float* __restrict__ kb,
                                                const float* __restrict__ vb,
                                                _Float16* __restrict__ Qh,
                                                _Float16* __restrict__ Kt,
                                                _Float16* __restrict__ Vt) {
    int blk  = blockIdx.x;
    int b    = blk >> 8;                  // / (S_/16)
    int s0   = (blk & 255) * 16;
    int wave = threadIdx.x >> 5;
    int lane = threadIdx.x & 31;
    int e0   = wave * 16;

    const float*    xrow = x  + ((size_t)b * S_ + s0) * D_;
    const _Float16* Wq   = Wh + (size_t)e0 * D_;
    const _Float16* Wk   = Wq + (size_t)E_ * D_;
    const _Float16* Wv   = Wk + (size_t)E_ * D_;

    v8f aq = {}, ak = {}, av = {};
    for (int k0 = 0; k0 < D_; k0 += 32) {
        v16h a  = load_a_f32cvt(xrow, D_, k0);
        v16h bq = load_b_f16(Wq, D_, k0);
        v16h bk = load_b_f16(Wk, D_, k0);
        v16h bv = load_b_f16(Wv, D_, k0);
        aq = __builtin_amdgcn_wmma_f32_16x16x32_f16(false, a, false, bq, (short)0, aq, false, false);
        ak = __builtin_amdgcn_wmma_f32_16x16x32_f16(false, a, false, bk, (short)0, ak, false, false);
        av = __builtin_amdgcn_wmma_f32_16x16x32_f16(false, a, false, bv, (short)0, av, false, false);
    }

    int ncol = lane & 15;
    int moff = (lane >> 4) * 8;           // C/D layout: lanes16-31 hold M=r+8
    float bqs = qb[e0 + ncol], bks = kb[e0 + ncol], bvs = vb[e0 + ncol];

    // Q row-major (strided b16 stores)
    #pragma unroll
    for (int r = 0; r < 8; ++r)
        Qh[((size_t)b * S_ + s0 + r + moff) * E_ + e0 + ncol] = (_Float16)(aq[r] + bqs);

    // K,V transposed: per-lane 8 consecutive seq positions -> one b128 store
    h8 kp, vp;
    #pragma unroll
    for (int r = 0; r < 8; ++r) { kp[r] = (_Float16)(ak[r] + bks); vp[r] = (_Float16)(av[r] + bvs); }
    size_t tb = ((size_t)b * E_ + e0 + ncol) * S_ + s0 + moff;
    *(h8*)(Kt + tb) = kp;
    *(h8*)(Vt + tb) = vp;
}

// ---- Phase 2: M = K^T V  (per batch 128x128, K-dim 4096, k-split 16) ------
__global__ __launch_bounds__(256) void kv_outer(const _Float16* __restrict__ Kt,
                                                const _Float16* __restrict__ Vt,
                                                float* __restrict__ M) {
    int b     = blockIdx.x >> 4;
    int ks    = blockIdx.x & 15;
    int wave  = threadIdx.x >> 5;
    int lane  = threadIdx.x & 31;
    int mbase = wave * 16;

    const _Float16* Krows = Kt + ((size_t)b * E_ + mbase) * S_;  // A rows = K^T rows
    const _Float16* Vrows = Vt + (size_t)b * E_ * S_;            // B^T rows = V cols

    v8f acc[8];
    #pragma unroll
    for (int t = 0; t < 8; ++t) acc[t] = (v8f){};

    int kbeg = ks * (S_ / 16);
    for (int k0 = kbeg; k0 < kbeg + S_ / 16; k0 += 32) {
        v16h a = load_a_f16(Krows, S_, k0);
        #pragma unroll
        for (int nt = 0; nt < 8; ++nt) {
            v16h bf = load_b_f16(Vrows + (size_t)nt * 16 * S_, S_, k0);
            acc[nt] = __builtin_amdgcn_wmma_f32_16x16x32_f16(false, a, false, bf, (short)0, acc[nt], false, false);
        }
    }

    int ncol = lane & 15, moff = (lane >> 4) * 8;
    #pragma unroll
    for (int nt = 0; nt < 8; ++nt)
        #pragma unroll
        for (int r = 0; r < 8; ++r)
            unsafeAtomicAdd(&M[(size_t)b * E_ * E_ + (mbase + r + moff) * E_ + nt * 16 + ncol],
                            acc[nt][r]);
}

// ---- Phase 2.5: Mt[b][n][k] = f16( M[b][k][n] * scale ) -------------------
__global__ __launch_bounds__(256) void scale_mt(const float* __restrict__ M,
                                                _Float16* __restrict__ Mt) {
    int idx = blockIdx.x * 256 + threadIdx.x;   // 0 .. B_*E_*E_-1
    int b = idx >> 14, r = idx & 16383;
    int n = r >> 7,    k = r & 127;
    Mt[idx] = (_Float16)(M[(b << 14) + (k << 7) + n] * SCALE_);
}

// ---- Phase 3: out = Q @ (M*scale), f32 output -----------------------------
__global__ __launch_bounds__(256) void out_proj(const _Float16* __restrict__ Qh,
                                                const _Float16* __restrict__ Mt,
                                                float* __restrict__ out) {
    int blk  = blockIdx.x;
    int b    = blk >> 8;
    int s0   = (blk & 255) * 16;
    int wave = threadIdx.x >> 5;
    int lane = threadIdx.x & 31;
    int e0   = wave * 16;

    const _Float16* Qrows = Qh + ((size_t)b * S_ + s0) * E_;
    const _Float16* Mrows = Mt + (size_t)b * E_ * E_ + (size_t)e0 * E_;

    v8f acc = {};
    #pragma unroll
    for (int k0 = 0; k0 < E_; k0 += 32) {
        v16h a  = load_a_f16(Qrows, E_, k0);
        v16h bf = load_b_f16(Mrows, E_, k0);
        acc = __builtin_amdgcn_wmma_f32_16x16x32_f16(false, a, false, bf, (short)0, acc, false, false);
    }

    int ncol = lane & 15, moff = (lane >> 4) * 8;
    #pragma unroll
    for (int r = 0; r < 8; ++r)
        out[((size_t)b * S_ + s0 + r + moff) * E_ + e0 + ncol] = acc[r];
}

// ---------------------------------------------------------------------------
extern "C" void kernel_launch(void* const* d_in, const int* in_sizes, int n_in,
                              void* d_out, int out_size, void* d_ws, size_t ws_size,
                              hipStream_t stream) {
    const float* x  = (const float*)d_in[0];
    const float* qW = (const float*)d_in[1];
    const float* qb = (const float*)d_in[2];
    const float* kW = (const float*)d_in[3];
    const float* kb = (const float*)d_in[4];
    const float* vW = (const float*)d_in[5];
    const float* vb = (const float*)d_in[6];
    float* out = (float*)d_out;

    // workspace layout (bytes)
    char* ws = (char*)d_ws;
    _Float16* Wh = (_Float16*)(ws);                                  // 3*E*D f16   = 768 KB
    _Float16* Qh = (_Float16*)(ws + 786432);                         // B*S*E f16   = 2 MB
    _Float16* Kt = (_Float16*)(ws + 786432 + 2097152);               // B*E*S f16   = 2 MB
    _Float16* Vt = (_Float16*)(ws + 786432 + 2 * 2097152);           // B*E*S f16   = 2 MB
    float*    M  = (float*)   (ws + 786432 + 3 * 2097152);           // B*E*E f32   = 128 KB
    _Float16* Mt = (_Float16*)(ws + 786432 + 3 * 2097152 + 131072);  // B*E*E f16   = 64 KB

    conv_w  <<<(E_ * D_) / 256, 256, 0, stream>>>(qW, kW, vW, Wh);
    zero_m  <<<(B_ * E_ * E_) / 256, 256, 0, stream>>>(M);
    qkv_proj<<<(B_ * S_) / 16, 256, 0, stream>>>(x, Wh, qb, kb, vb, Qh, Kt, Vt);
    kv_outer<<<B_ * 16, 256, 0, stream>>>(Kt, Vt, M);
    scale_mt<<<(B_ * E_ * E_) / 256, 256, 0, stream>>>(M, Mt);
    out_proj<<<(B_ * S_) / 16, 256, 0, stream>>>(Qh, Mt, out);
}
